// SlidingWindowSelfAttention_77206332113045
// MI455X (gfx1250) — matrix operand order repaired and checked by reading, not compiled, not run on hardware
//
#include <hip/hip_runtime.h>

// ============================================================================
// SlidingWindow (blocked) MHA for MI455X / gfx1250, wave32, WMMA f16.
// B=4, T=4096, C=1024, H=16 (hd=64), G=16 (kd=64), WINDOW=256, rope base 1e4.
// Bandwidth-bound (~155 GFLOP vs ~0.5GB traffic): f16 data path, fp32 accum.
// GEMMs: double-buffered LDS staging via GLOBAL_LOAD_ASYNC_TO_LDS_B128
// (ASYNCcnt), one barrier per K-step, v_wmma_f32_16x16x32_f16 compute.
// Workspace layout (bytes):           offset        size
//   xh    f16 [16384,1024]                0      33,554,432
//   WinT  f16 [3072,1024]  (W_in^T)  33,554,432   6,291,456
//   WoutT f16 [1024,1024]  (W_out^T) 39,845,888   2,097,152
//   qkvh  f16 [16384,3072]           41,943,040 100,663,296
//   qh    f16 [4,16,4096,64]        142,606,336  33,554,432
//   kh    f16 [4,16,4096,64]        176,160,768  33,554,432
//   vh    f16 [4,16,4096,64]        209,715,200  33,554,432
//   yh    f16 [16384,1024]          243,269,632  33,554,432   (end 276,824,064)
// ============================================================================

typedef __attribute__((ext_vector_type(16))) _Float16 v16h;
typedef __attribute__((ext_vector_type(8)))  float    v8f;

union FragU { uint4 q[2]; v16h h; };

__device__ __forceinline__ v8f wmma16x16x32(v16h a, v16h b, v8f c) {
  // D = A(16x32 f16) x B(32x16 f16) + C(16x16 f32)
  return __builtin_amdgcn_wmma_f32_16x16x32_f16(false, a, false, b, (short)0, c,
                                                false, false);
}

// Async copy: 16B per lane, global -> LDS, tracked by ASYNCcnt.
// lds_off: per-lane LDS byte offset (16B aligned); gaddr: per-lane global addr.
__device__ __forceinline__ void async_load_b128(unsigned lds_off,
                                                const _Float16* gaddr) {
  asm volatile("global_load_async_to_lds_b128 %0, %1, off"
               :: "v"(lds_off), "v"(gaddr)
               : "memory");
}
__device__ __forceinline__ void wait_async0() {
  asm volatile("s_wait_asynccnt 0x0" ::: "memory");
}

// A fragment (16x32 f16). Lane: M row = lane&15, half = lane>>4.
// Element i: K = (i>>3)*16 + half*8 + (i&7)  -> two contiguous 8-half chunks.
// Requires (row*stride + k0 + half*8) to be 8-half (16B) aligned.
__device__ __forceinline__ v16h frag_a(const _Float16* p, int stride_h, int row,
                                       int k0, int lhi) {
  FragU f;
  const _Float16* r = p + (size_t)row * stride_h + k0 + lhi * 8;
  f.q[0] = *(const uint4*)(r);
  f.q[1] = *(const uint4*)(r + 16);
  return f.h;
}

// B fragment (32x16 f16) taken from Bt stored [N][K] (row = output col n).
// Lane: N col = lane&15, half = lane>>4. Element i: K = (i>>3)*16 + half*8 + (i&7)
// -> two contiguous 8-half chunks.
__device__ __forceinline__ v16h frag_b(const _Float16* p, int stride_h, int col,
                                       int k0, int lhi) {
  FragU f;
  const _Float16* r = p + (size_t)col * stride_h + k0 + lhi * 8;
  f.q[0] = *(const uint4*)(r);
  f.q[1] = *(const uint4*)(r + 16);
  return f.h;
}

// ---------------------------------------------------------------------------
// fp32 -> f16 convert, 4 elements / thread
__global__ __launch_bounds__(256) void cvt_f32_to_f16(
    const float* __restrict__ src, _Float16* __restrict__ dst, int n4) {
  int i = blockIdx.x * blockDim.x + threadIdx.x;
  if (i >= n4) return;
  float4 v = ((const float4*)src)[i];
  union { _Float16 h[4]; uint2 u; } o;
  o.h[0] = (_Float16)v.x; o.h[1] = (_Float16)v.y;
  o.h[2] = (_Float16)v.z; o.h[3] = (_Float16)v.w;
  ((uint2*)dst)[i] = o.u;
}

// transpose + convert: src f32 [K][N] -> dst f16 [N][K] (coalesced writes)
__global__ __launch_bounds__(256) void transpose_cvt(
    const float* __restrict__ src, _Float16* __restrict__ dst, int K, int N) {
  int idx = blockIdx.x * blockDim.x + threadIdx.x;
  if (idx >= K * N) return;
  int k = idx % K, n = idx / K;
  dst[(size_t)n * K + k] = (_Float16)src[(size_t)k * N + n];
}

// ---------------------------------------------------------------------------
// WMMA GEMM: C[M,N] = A[M,K](f16, row major) * Bt[N,K]^T (f16) + bias[N]
// block tile 64x64, 4 waves (128 thr), K-step 32, M,N%64==0, K%32==0.
// Double-buffered async (global->LDS) staging: ASYNCcnt + 1 barrier / step.
__shared__ _Float16 gemm_As[2][64][40];  // pad 32->40 halves: 80B row, 16B ok
__shared__ _Float16 gemm_Bs[2][64][40];

template <bool OUT_F32>
__global__ __launch_bounds__(128) void gemm_wmma(
    const _Float16* __restrict__ A, const _Float16* __restrict__ Bt,
    const float* __restrict__ bias, void* __restrict__ Cout,
    int M, int N, int K) {
  int n0 = blockIdx.x * 64, m0 = blockIdx.y * 64;
  int t = threadIdx.x, wave = t >> 5, lane = t & 31;
  int llo = lane & 15, lhi = lane >> 4;
  int lrow = t >> 1, lcol = (t & 1) * 16;   // 2 threads/row, 16 halves each
  const _Float16* Ag = A  + (size_t)(m0 + lrow) * K + lcol;
  const _Float16* Bg = Bt + (size_t)(n0 + lrow) * K + lcol;
  unsigned aoff[2], boff[2];
#pragma unroll
  for (int p = 0; p < 2; p++) {
    aoff[p] = (unsigned)(uintptr_t)&gemm_As[p][lrow][lcol];
    boff[p] = (unsigned)(uintptr_t)&gemm_Bs[p][lrow][lcol];
  }
  v8f zero = {0.f, 0.f, 0.f, 0.f, 0.f, 0.f, 0.f, 0.f};
  v8f acc[4] = {zero, zero, zero, zero};

  // Prologue: stage K-tile 0 into buffer 0 (32B per thread = 2 x b128).
  async_load_b128(aoff[0],      Ag);
  async_load_b128(aoff[0] + 16, Ag + 8);
  async_load_b128(boff[0],      Bg);
  async_load_b128(boff[0] + 16, Bg + 8);

  int buf = 0;
  for (int kt = 0; kt < K; kt += 32) {
    wait_async0();        // this wave's async stores to LDS are complete
    __syncthreads();      // all waves: buf ready, previous compute finished
    int nbuf = buf ^ 1;
    if (kt + 32 < K) {    // stage next K-tile into the other buffer
      async_load_b128(aoff[nbuf],      Ag + kt + 32);
      async_load_b128(aoff[nbuf] + 16, Ag + kt + 40);
      async_load_b128(boff[nbuf],      Bg + kt + 32);
      async_load_b128(boff[nbuf] + 16, Bg + kt + 40);
    }
    v16h af = frag_a(&gemm_As[buf][0][0], 40, wave * 16 + llo, 0, lhi);
#pragma unroll
    for (int nt = 0; nt < 4; nt++) {
      v16h bf = frag_b(&gemm_Bs[buf][0][0], 40, nt * 16 + llo, 0, lhi);
      acc[nt] = wmma16x16x32(af, bf, acc[nt]);
    }
    buf = nbuf;
  }
  // D layout: vgpr r -> row = lhi*8 + r, col = llo
#pragma unroll
  for (int nt = 0; nt < 4; nt++) {
#pragma unroll
    for (int r = 0; r < 8; r++) {
      int m = m0 + wave * 16 + lhi * 8 + r;
      int n = n0 + nt * 16 + llo;
      float v = acc[nt][r] + bias[n];
      if constexpr (OUT_F32) ((float*)Cout)[(size_t)m * N + n] = v;
      else ((_Float16*)Cout)[(size_t)m * N + n] = (_Float16)v;
    }
  }
}

// ---------------------------------------------------------------------------
// RoPE on q,k + repack q/k/v: qkvh[b*T+t][3072] -> {q,k,v}h[b][h][t][64].
// One thread per (b,t,h,pair i); i in [0,32).
__global__ __launch_bounds__(256) void rope_pack(
    const _Float16* __restrict__ qkv, _Float16* __restrict__ qh,
    _Float16* __restrict__ kh, _Float16* __restrict__ vh) {
  int idx = blockIdx.x * blockDim.x + threadIdx.x;   // 2^23 threads
  int i  = idx & 31;
  int h  = (idx >> 5) & 15;
  int tt = (idx >> 9) & 4095;
  int b  = idx >> 21;
  size_t src = ((size_t)(b * 4096 + tt)) * 3072 + h * 64 + 2 * i;
  // inv_freq = 10000^(-2i/64)
  float inv_freq = __expf(-(float)(2 * i) * (9.210340371976184f / 64.0f));
  float ang = (float)tt * inv_freq;
  float s, c;
  __sincosf(ang, &s, &c);
  union { uint u; _Float16 h2[2]; } iq, ik, iv, oq, ok;
  iq.u = *(const uint*)(qkv + src);
  ik.u = *(const uint*)(qkv + src + 1024);
  iv.u = *(const uint*)(qkv + src + 2048);
  float q1 = (float)iq.h2[0], q2 = (float)iq.h2[1];
  float k1 = (float)ik.h2[0], k2 = (float)ik.h2[1];
  oq.h2[0] = (_Float16)(q1 * c - q2 * s);
  oq.h2[1] = (_Float16)(q1 * s + q2 * c);
  ok.h2[0] = (_Float16)(k1 * c - k2 * s);
  ok.h2[1] = (_Float16)(k1 * s + k2 * c);
  size_t dst = (((size_t)(b * 16 + h)) * 4096 + tt) * 64 + 2 * i;
  *(uint*)(qh + dst) = oq.u;
  *(uint*)(kh + dst) = ok.u;
  *(uint*)(vh + dst) = iv.u;
}

// ---------------------------------------------------------------------------
// Blocked attention: one block per (b, h, window, 64-row q strip).
// 256 threads = 8 waves. S=Q*K^T (WMMA, frags straight from global, L2-hot),
// softmax (f32 LDS, wave reductions), P overlaid f16 in-place, O=P*V (WMMA,
// V transposed into LDS). Dynamic LDS = 64*256*4 + 64*264*2 = 99,328 B.
__global__ __launch_bounds__(256) void attn_win(
    const _Float16* __restrict__ qh, const _Float16* __restrict__ kh,
    const _Float16* __restrict__ vh, _Float16* __restrict__ yh) {
  extern __shared__ char smem[];
  float*    Sf = (float*)smem;                    // [64][256] f32
  _Float16* Ps = (_Float16*)smem;                 // overlay: [64] rows, stride 512h
  _Float16* Vt = (_Float16*)(smem + 64 * 256 * 4);// [64][264] f16 (d-major)

  int bid   = blockIdx.x;
  int strip = bid & 3;
  int win   = (bid >> 2) & 15;
  int h     = (bid >> 6) & 15;
  int b     = bid >> 10;
  size_t headbase = ((size_t)(b * 16 + h)) * 4096 * 64;
  int q0  = win * 256 + strip * 64;   // first query token of this strip
  int k0g = win * 256;                // first key token of this window

  int t = threadIdx.x, wave = t >> 5, lane = t & 31;
  int llo = lane & 15, lhi = lane >> 4;

  // Stage V transposed: Vt[d][keypos], one key row per thread.
  {
    union { uint4 q[8]; _Float16 hh[64]; } vr;
    const uint4* Vg = (const uint4*)(vh + headbase + (size_t)(k0g + t) * 64);
#pragma unroll
    for (int j = 0; j < 8; j++) vr.q[j] = Vg[j];
#pragma unroll
    for (int d = 0; d < 64; d++) Vt[d * 264 + t] = vr.hh[d];
  }

  // S = Q K^T, scaled. Wave: q-tile qt = wave>>1, k-tiles kb..kb+7.
  int qt = wave >> 1;
  int kb = (wave & 1) * 8;
  v8f zero = {0.f, 0.f, 0.f, 0.f, 0.f, 0.f, 0.f, 0.f};
  v8f acc[8] = {zero, zero, zero, zero, zero, zero, zero, zero};
  const _Float16* Qg = qh + headbase + (size_t)q0 * 64;   // [64][64] row major
  const _Float16* Kg = kh + headbase + (size_t)k0g * 64;  // [256][64] row major
#pragma unroll
  for (int kk = 0; kk < 64; kk += 32) {
    v16h af = frag_a(Qg, 64, qt * 16 + llo, kk, lhi);
#pragma unroll
    for (int j = 0; j < 8; j++) {
      v16h bf = frag_b(Kg, 64, (kb + j) * 16 + llo, kk, lhi);
      acc[j] = wmma16x16x32(af, bf, acc[j]);
    }
  }
  const float scale = 0.125f;  // 1/sqrt(64)
#pragma unroll
  for (int j = 0; j < 8; j++)
#pragma unroll
    for (int r = 0; r < 8; r++)
      Sf[(qt * 16 + lhi * 8 + r) * 256 + (kb + j) * 16 + llo] = acc[j][r] * scale;
  __syncthreads();

  // Softmax: wave w owns rows w*8..w*8+7; lanes span 256 cols.
  for (int rr = 0; rr < 8; rr++) {
    int row = wave * 8 + rr;
    float vv[8];
    float mx = -3.0e38f;
#pragma unroll
    for (int j = 0; j < 8; j++) {
      vv[j] = Sf[row * 256 + lane + j * 32];
      mx = fmaxf(mx, vv[j]);
    }
#pragma unroll
    for (int off = 16; off >= 1; off >>= 1) mx = fmaxf(mx, __shfl_xor(mx, off));
    float sum = 0.f;
#pragma unroll
    for (int j = 0; j < 8; j++) { vv[j] = __expf(vv[j] - mx); sum += vv[j]; }
#pragma unroll
    for (int off = 16; off >= 1; off >>= 1) sum += __shfl_xor(sum, off);
    float invs = 1.0f / sum;
    // overlay P(f16) into this row's bytes (row owned solely by this wave)
#pragma unroll
    for (int j = 0; j < 8; j++)
      Ps[row * 512 + lane + j * 32] = (_Float16)(vv[j] * invs);
  }
  __syncthreads();

  // O = P V. Wave: q-tile qt, n-tiles nt0, nt0+1.
  int nt0 = (wave & 1) * 2;
  v8f oacc[2] = {zero, zero};
#pragma unroll
  for (int kk = 0; kk < 256; kk += 32) {
    v16h af = frag_a(Ps, 512, qt * 16 + llo, kk, lhi);
#pragma unroll
    for (int j = 0; j < 2; j++) {
      v16h bf = frag_b(Vt, 264, (nt0 + j) * 16 + llo, kk, lhi);
      oacc[j] = wmma16x16x32(af, bf, oacc[j]);
    }
  }
  // write y[b][t][h*64+d] (f16 for final GEMM)
#pragma unroll
  for (int j = 0; j < 2; j++)
#pragma unroll
    for (int r = 0; r < 8; r++) {
      int mrow = qt * 16 + lhi * 8 + r;
      int d = (nt0 + j) * 16 + llo;
      yh[(size_t)(b * 4096 + q0 + mrow) * 1024 + h * 64 + d] = (_Float16)oacc[j][r];
    }
}

// ---------------------------------------------------------------------------
extern "C" void kernel_launch(void* const* d_in, const int* in_sizes, int n_in,
                              void* d_out, int out_size, void* d_ws,
                              size_t ws_size, hipStream_t stream) {
  (void)in_sizes; (void)n_in; (void)out_size; (void)ws_size;
  const float* x     = (const float*)d_in[0];
  const float* W_in  = (const float*)d_in[1];
  const float* b_in  = (const float*)d_in[2];
  const float* W_out = (const float*)d_in[3];
  const float* b_out = (const float*)d_in[4];
  float* out = (float*)d_out;

  char* ws = (char*)d_ws;
  _Float16* xh    = (_Float16*)(ws);
  _Float16* WinT  = (_Float16*)(ws + 33554432);
  _Float16* WoutT = (_Float16*)(ws + 39845888);
  _Float16* qkvh  = (_Float16*)(ws + 41943040);
  _Float16* qh    = (_Float16*)(ws + 142606336);
  _Float16* kh    = (_Float16*)(ws + 176160768);
  _Float16* vh    = (_Float16*)(ws + 209715200);
  _Float16* yh    = (_Float16*)(ws + 243269632);

  const int M = 16384, C = 1024, N3 = 3072;

  // 1) x -> f16 ; W_in, W_out -> transposed f16
  cvt_f32_to_f16<<<(M * C / 4 + 255) / 256, 256, 0, stream>>>(x, xh, M * C / 4);
  transpose_cvt<<<(C * N3 + 255) / 256, 256, 0, stream>>>(W_in, WinT, C, N3);
  transpose_cvt<<<(C * C + 255) / 256, 256, 0, stream>>>(W_out, WoutT, C, C);

  // 2) qkv = x @ W_in + b_in   (f16 out)
  gemm_wmma<false><<<dim3(N3 / 64, M / 64), 128, 0, stream>>>(
      xh, WinT, b_in, (void*)qkvh, M, N3, C);

  // 3) RoPE + repack to [b][h][t][64]
  rope_pack<<<(4 * 4096 * 16 * 32) / 256, 256, 0, stream>>>(qkvh, qh, kh, vh);

  // 4) blocked attention: 4 (B) * 16 (H) * 16 (win) * 4 (strips) = 4096 blocks
  attn_win<<<4096, 256, 64 * 256 * 4 + 64 * 264 * 2, stream>>>(qh, kh, vh, yh);

  // 5) out = y @ W_out + b_out  (f32 out)
  gemm_wmma<true><<<dim3(C / 64, M / 64), 128, 0, stream>>>(
      yh, WoutT, b_out, (void*)out, M, C, C);
}
